// AFTFull_18133351923892
// MI455X (gfx1250) — matrix operand order repaired and checked by reading, not compiled, not run on hardware
//
#include <hip/hip_runtime.h>
#include <hip/hip_bf16.h>

typedef __bf16 bf16_t;
typedef __attribute__((ext_vector_type(16))) __bf16 v16bf;
typedef __attribute__((ext_vector_type(8)))  __bf16 v8bf;
typedef __attribute__((ext_vector_type(8)))  float  v8f;

typedef __attribute__((vector_size(16))) int i32x4;
typedef __attribute__((vector_size(8)))  int i32x2;

#define BB 16
#define TT 1024
#define DD 1024
#define HH 16
#define HD 64
#define LPAD 40   // padded LDS row stride in bf16 (20 dwords -> conflict-free)

// ---------------- async copy to LDS (CDNA5, ASYNCcnt-tracked) ----------------
#if defined(__has_builtin)
#if __has_builtin(__builtin_amdgcn_global_load_async_to_lds_b128) && \
    __has_builtin(__builtin_amdgcn_global_load_async_to_lds_b64)
#define HAVE_ASYNC_LDS 1
#endif
#if __has_builtin(__builtin_amdgcn_sched_barrier)
#define SCHED_FENCE() __builtin_amdgcn_sched_barrier(0)
#endif
#endif
#ifndef SCHED_FENCE
#define SCHED_FENCE()
#endif

#ifdef HAVE_ASYNC_LDS
__device__ __forceinline__ void cp_async_b128(void* dst_lds, const void* src_g) {
    __builtin_amdgcn_global_load_async_to_lds_b128(
        (__attribute__((address_space(1))) i32x4*)(uintptr_t)src_g,
        (__attribute__((address_space(3))) i32x4*)(uint32_t)(uintptr_t)dst_lds,
        0, 0);
}
__device__ __forceinline__ void cp_async_b64(void* dst_lds, const void* src_g) {
    __builtin_amdgcn_global_load_async_to_lds_b64(
        (__attribute__((address_space(1))) i32x2*)(uintptr_t)src_g,
        (__attribute__((address_space(3))) i32x2*)(uint32_t)(uintptr_t)dst_lds,
        0, 0);
}
#define WAIT_ASYNC(n) asm volatile("s_wait_asynccnt " #n ::: "memory")
#endif

// ---------------- fragment loaders (ISA 7.12.2 layouts) ----------------
// A (16x32 bf16, M x K): lanes 0-15 = rows, K chunks {0..7,16..23};
// lanes 16-31 = rows, K chunks {8..15,24..31}.
__device__ __forceinline__ v16bf load_frag_a(const bf16_t* __restrict__ base,
                                             int stride, int lane) {
    int r  = lane & 15;
    int hi = lane >> 4;
    const bf16_t* p = base + (size_t)r * stride + hi * 8;
    union { v16bf v; v8bf h[2]; } u;
    u.h[0] = *(const v8bf*)(p);
    u.h[1] = *(const v8bf*)(p + 16);
    return u.v;
}

// B (32x16 bf16, K x N) from [N][K]-contiguous storage (global or LDS):
// lane n(0-15): col n, K {0..7,16..23}; lanes 16-31: col n, K {8..15,24..31}.
// Built from two 16B halves so 16B alignment suffices (padded LDS rows).
__device__ __forceinline__ v16bf load_frag_b(const bf16_t* base,
                                             int stride, int lane) {
    int n  = lane & 15;
    int k0 = (lane >> 4) << 3;
    const bf16_t* p = base + (size_t)n * stride + k0;
    union { v16bf v; v8bf h[2]; } u;
    u.h[0] = *(const v8bf*)(p);
    u.h[1] = *(const v8bf*)(p + 16);
    return u.v;
}

__device__ __forceinline__ v8f wmma_bf16(v16bf a, v16bf b, v8f c) {
    return __builtin_amdgcn_wmma_f32_16x16x32_bf16(false, a, false, b,
                                                   (short)0, c, false, false);
}

// ---------------- phase 0: conversions ----------------
__global__ void cvt_f32_bf16(const float* __restrict__ x, bf16_t* __restrict__ y, int n4) {
    int i = blockIdx.x * blockDim.x + threadIdx.x;
    if (i < n4) {
        float4 v = ((const float4*)x)[i];
        bf16_t* o = y + (size_t)i * 4;
        o[0] = (bf16_t)v.x; o[1] = (bf16_t)v.y; o[2] = (bf16_t)v.z; o[3] = (bf16_t)v.w;
    }
}

__global__ void exp_f32_bf16(const float* __restrict__ x, bf16_t* __restrict__ y, int n4) {
    int i = blockIdx.x * blockDim.x + threadIdx.x;
    if (i < n4) {
        float4 v = ((const float4*)x)[i];
        bf16_t* o = y + (size_t)i * 4;
        o[0] = (bf16_t)__expf(v.x); o[1] = (bf16_t)__expf(v.y);
        o[2] = (bf16_t)__expf(v.z); o[3] = (bf16_t)__expf(v.w);
    }
}

// ---------------- phase 1: fused q/k/v projections ----------------
// 256-thread block = 8 waves, all sharing one 32-wide output-column group.
// Weight tiles (3 proj x 32n x 32k bf16 = 6KB/step) staged in LDS once per
// block (async double-buffered); each wave does its own 16-row t-tile.
__global__ void __launch_bounds__(256)
proj_qkv(const bf16_t* __restrict__ xb,
         const bf16_t* __restrict__ wq,
         const bf16_t* __restrict__ wk,
         const bf16_t* __restrict__ wv,
         bf16_t* __restrict__ sqT,
         bf16_t* __restrict__ ekT,
         bf16_t* __restrict__ ekvT) {
    __shared__ bf16_t wst[2][3][32 * LPAD];

    int tid  = threadIdx.x;
    int lane = tid & 31;
    int w    = tid >> 5;
    int dd   = blockIdx.x & 31;        // 32 column groups of 32
    int tg   = blockIdx.x >> 5;        // 128 t-groups of 8 tiles
    int t0   = (tg * 8 + w) * 16;      // global row in [0, B*T)
    int d0   = dd * 32;

    const bf16_t* wp[3] = {wq, wk, wv};

    // stage weight tiles for k-step kk into buffer `buf`
    // 3 tiles x 32 rows x 32 bf16 = 768 b64 chunks, 3 per thread (uniform)
    auto stage = [&](int kk, int buf) {
        int k0 = kk * 32;
#pragma unroll
        for (int it = 0; it < 3; ++it) {
            int idx  = tid + it * 256;
            int proj = idx >> 8;
            int c    = idx & 255;
            int row  = c >> 3;
            int ch   = c & 7;              // 8B chunk within 32-elem row
            const bf16_t* src = wp[proj] + (size_t)(d0 + row) * DD + k0 + ch * 4;
            bf16_t* dst = &wst[buf][proj][row * LPAD + ch * 4];
#ifdef HAVE_ASYNC_LDS
            cp_async_b64(dst, src);
#else
            *(uint2*)dst = *(const uint2*)src;
#endif
        }
    };

    v8f acc[3][2] = {};

    auto compute = [&](int kk, int buf) {
        v16bf a = load_frag_a(xb + (size_t)t0 * DD + kk * 32, DD, lane);
        // hoist all B fragments; sched fence keeps all ds_loads before the
        // WMMA chain so fragments get distinct registers (one wait, 6 WMMAs)
        v16bf bw[3][2];
#pragma unroll
        for (int p = 0; p < 3; ++p)
#pragma unroll
            for (int ni = 0; ni < 2; ++ni)
                bw[p][ni] = load_frag_b(&wst[buf][p][ni * 16 * LPAD], LPAD, lane);
        SCHED_FENCE();
#pragma unroll
        for (int p = 0; p < 3; ++p)
#pragma unroll
            for (int ni = 0; ni < 2; ++ni)
                acc[p][ni] = wmma_bf16(a, bw[p][ni], acc[p][ni]);
    };

#ifdef HAVE_ASYNC_LDS
    stage(0, 0);
    stage(1, 1);
    for (int kk = 0; kk < DD / 32; ++kk) {
        if (kk < DD / 32 - 1) { WAIT_ASYNC(0x3); } else { WAIT_ASYNC(0x0); }
        __syncthreads();
        compute(kk, kk & 1);
        __syncthreads();
        if (kk + 2 < DD / 32) stage(kk + 2, kk & 1);
    }
#else
    for (int kk = 0; kk < DD / 32; ++kk) {
        __syncthreads();
        stage(kk, 0);
        __syncthreads();
        compute(kk, 0);
    }
#endif

    int b  = t0 >> 10;
    int tl = t0 & (TT - 1);
    int hi = lane >> 4;
    int tb = tl + hi * 8;
#pragma unroll
    for (int ni = 0; ni < 2; ++ni) {
        int dglob = d0 + ni * 16 + (lane & 15);
        int h = dglob >> 6, dh = dglob & 63;
        v8bf sq, ek, ekv;
#pragma unroll
        for (int j = 0; j < 8; ++j) {
            float e = __expf(acc[1][ni][j]);
            sq[j]  = (bf16_t)(1.0f / (1.0f + __expf(-acc[0][ni][j])));
            ek[j]  = (bf16_t)e;
            ekv[j] = (bf16_t)(e * acc[2][ni][j]);
        }
        *(v8bf*)(sqT + ((size_t)b * DD + dglob) * TT + tb) = sq;
        size_t hb = (((size_t)b * HH + h) * HD + dh) * TT + tb;
        *(v8bf*)(ekT + hb)  = ek;
        *(v8bf*)(ekvT + hb) = ekv;
    }
}

// ---------------- phase 2: num/den GEMMs + epilogue ----------------
// 256-thread block = 8 waves, all on the same (b,h); ekv/ek tiles
// (2 x 64d x 32s bf16 = 8KB/step) staged in LDS once per block.
__global__ void __launch_bounds__(256)
aft_core(const bf16_t* __restrict__ ewb,
         const bf16_t* __restrict__ ekT,
         const bf16_t* __restrict__ ekvT,
         const bf16_t* __restrict__ sqT,
         float* __restrict__ out) {
    __shared__ bf16_t est[2][2][HD * LPAD];
    __shared__ float  smem[8][16][64];

    int tid  = threadIdx.x;
    int lane = tid & 31;
    int w    = tid >> 5;
    int tg   = blockIdx.x & 7;
    int h    = (blockIdx.x >> 3) & 15;
    int b    = blockIdx.x >> 7;
    int t0   = (tg * 8 + w) * 16;

    const bf16_t* ekvh = ekvT + (((size_t)b * HH + h) * HD) * TT;
    const bf16_t* ekh  = ekT  + (((size_t)b * HH + h) * HD) * TT;
    const bf16_t* srcs[2] = {ekvh, ekh};

    // 2 tiles x 64 rows x 32 bf16 = 512 b128 chunks, 2 per thread (uniform)
    auto stage = [&](int ss, int buf) {
        int s0 = ss * 32;
#pragma unroll
        for (int it = 0; it < 2; ++it) {
            int idx   = tid + it * 256;
            int which = idx >> 8;
            int c     = idx & 255;
            int row   = c >> 2;
            int ch    = c & 3;             // 16B chunk within 32-elem row
            const bf16_t* src = srcs[which] + (size_t)row * TT + s0 + ch * 8;
            bf16_t* dst = &est[buf][which][row * LPAD + ch * 8];
#ifdef HAVE_ASYNC_LDS
            cp_async_b128(dst, src);
#else
            *(uint4*)dst = *(const uint4*)src;
#endif
        }
    };

    v8f num[4] = {}, den[4] = {};

    auto compute = [&](int ss, int buf) {
        v16bf a = load_frag_a(ewb + (size_t)t0 * TT + ss * 32, TT, lane);
        // hoist all 8 B fragments; sched fence -> one ds_load clause,
        // single wait, then 8 WMMAs back-to-back
        v16bf bn[4], bd[4];
#pragma unroll
        for (int ni = 0; ni < 4; ++ni) {
            bn[ni] = load_frag_b(&est[buf][0][ni * 16 * LPAD], LPAD, lane);
            bd[ni] = load_frag_b(&est[buf][1][ni * 16 * LPAD], LPAD, lane);
        }
        SCHED_FENCE();
#pragma unroll
        for (int ni = 0; ni < 4; ++ni) {
            num[ni] = wmma_bf16(a, bn[ni], num[ni]);
            den[ni] = wmma_bf16(a, bd[ni], den[ni]);
        }
    };

#ifdef HAVE_ASYNC_LDS
    stage(0, 0);
    stage(1, 1);
    for (int ss = 0; ss < TT / 32; ++ss) {
        if (ss < TT / 32 - 1) { WAIT_ASYNC(0x2); } else { WAIT_ASYNC(0x0); }
        __syncthreads();
        compute(ss, ss & 1);
        __syncthreads();
        if (ss + 2 < TT / 32) stage(ss + 2, ss & 1);
    }
#else
    for (int ss = 0; ss < TT / 32; ++ss) {
        __syncthreads();
        stage(ss, 0);
        __syncthreads();
        compute(ss, 0);
    }
#endif

    int hi = lane >> 4;
    int n  = lane & 15;
#pragma unroll
    for (int ni = 0; ni < 4; ++ni) {
        int dglob = h * HD + ni * 16 + n;
        v8bf sq = *(const v8bf*)(sqT + ((size_t)b * DD + dglob) * TT + t0 + hi * 8);
#pragma unroll
        for (int j = 0; j < 8; ++j)
            smem[w][hi * 8 + j][ni * 16 + n] = (float)sq[j] * num[ni][j] / den[ni][j];
    }

    asm volatile("s_wait_dscnt 0x0" ::: "memory");

    // coalesced write-back: lane covers 32 consecutive floats (half a row)
    int row = lane >> 1, c0 = (lane & 1) * 32;
    float* op = out + ((size_t)b * TT + t0 + row) * DD + h * HD + c0;
#pragma unroll
    for (int jj = 0; jj < 8; ++jj)
        ((float4*)op)[jj] = *(const float4*)&smem[w][row][c0 + jj * 4];
}

// ---------------- launch ----------------
extern "C" void kernel_launch(void* const* d_in, const int* in_sizes, int n_in,
                              void* d_out, int out_size, void* d_ws, size_t ws_size,
                              hipStream_t stream) {
    const float* x     = (const float*)d_in[0];
    const float* Wq    = (const float*)d_in[1];
    const float* Wk    = (const float*)d_in[2];
    const float* Wv    = (const float*)d_in[3];
    const float* wbias = (const float*)d_in[4];
    float* out = (float*)d_out;

    char* ws = (char*)d_ws;
    const size_t NX = (size_t)BB * TT * DD;   // 16M
    const size_t NW = (size_t)DD * DD;        // 1M
    bf16_t* xb   = (bf16_t*)(ws);                         // 32 MB
    bf16_t* wqb  = (bf16_t*)(ws + NX * 2);                // 2 MB
    bf16_t* wkb  = (bf16_t*)(ws + NX * 2 + NW * 2);
    bf16_t* wvb  = (bf16_t*)(ws + NX * 2 + NW * 4);
    bf16_t* ewb  = (bf16_t*)(ws + NX * 2 + NW * 6);       // exp(wbias), 2 MB
    bf16_t* sqT  = (bf16_t*)(ws + NX * 2 + NW * 8);       // 32 MB
    bf16_t* ekT  = (bf16_t*)(ws + NX * 4 + NW * 8);       // 32 MB
    bf16_t* ekvT = (bf16_t*)(ws + NX * 6 + NW * 8);       // 32 MB

    cvt_f32_bf16<<<(int)(NX / 4 / 256), 256, 0, stream>>>(x,  xb,  (int)(NX / 4));
    cvt_f32_bf16<<<(int)(NW / 4 / 256), 256, 0, stream>>>(Wq, wqb, (int)(NW / 4));
    cvt_f32_bf16<<<(int)(NW / 4 / 256), 256, 0, stream>>>(Wk, wkb, (int)(NW / 4));
    cvt_f32_bf16<<<(int)(NW / 4 / 256), 256, 0, stream>>>(Wv, wvb, (int)(NW / 4));
    exp_f32_bf16<<<(int)(NW / 4 / 256), 256, 0, stream>>>(wbias, ewb, (int)(NW / 4));

    proj_qkv<<<4096, 256, 0, stream>>>(xb, wqb, wkb, wvb, sqT, ekT, ekvT);
    aft_core<<<2048, 256, 0, stream>>>(ewb, ekT, ekvT, sqT, out);
}
